// PointCloudCorrector_13408887898660
// MI455X (gfx1250) — matrix operand order repaired and checked by reading, not compiled, not run on hardware
//
#include <hip/hip_runtime.h>

typedef _Float16 v16h __attribute__((ext_vector_type(16)));
typedef _Float16 h8   __attribute__((ext_vector_type(8)));
typedef float    v8f  __attribute__((ext_vector_type(8)));

#define NB      4
#define NPTS    400000
#define CBEV    256
#define CC      64
#define HH      180
#define WW      180
#define SS      10
#define MAXI    64
#define NUMLOC  2560
#define NUMINST 256
#define TPR     12                 // 16-wide tiles per row (180 -> 12)
#define NTILES  (NB*HH*TPR)        // 8640
#define XTPR    6                  // 30-wide col tiles for the HWC transpose

// ---- workspace layout (float offsets into d_ws) ----
#define WS_SUM      0        // 64  (BN sum)
#define WS_SQ       64       // 64  (BN sumsq)
#define WS_CNT      128      // 2560
#define WS_CSUM     2688     // 7680
#define WS_SCALE    10368    // 64
#define WS_SHIFT    10432    // 64
#define WS_CENTROID 10496    // 7680
#define WS_LOCKEY   18176    // 163840 (uint keys)
#define WS_SHPKEY   182016   // 163840 (uint keys)
#define WS_LOCALS   345856   // 163840
#define WS_GLOBALS  509696   // 16384
#define WS_INITC    526080   // 768
#define WS_TGTC     526848   // 768
#define WS_BSWZ     527616   // 147456 halves (73728 floats): WMMA-swizzled f16 weights
#define WS_YHWC     601344   // 8294400 floats: conv output, HWC f32
#define WS_XH       8895744  // 33177600 halves (16588800 floats): input, HWC f16
#define WS_ZBUF     25484544 // 256 floats (512 halves) of zeros for OOB columns

// output regions (float offsets into d_out)
#define R_REG 1200000
#define R_EMB 2400000
#define R_TF  3200000
#define R_MOS 3217920

#define NEGKEY 0x007FFFFFu   // encoded -inf

__device__ __forceinline__ unsigned fkey(float f) {
  unsigned u = __float_as_uint(f);
  return (u & 0x80000000u) ? ~u : (u | 0x80000000u);
}
__device__ __forceinline__ float fdec(unsigned k) {
  return (k & 0x80000000u) ? __uint_as_float(k ^ 0x80000000u) : __uint_as_float(~k);
}

// ---------------- kernel 0: init accumulators + zero pad buffer ----------------
__global__ void pcc_init(float* wsf, unsigned* wsu) {
  int i = blockIdx.x * blockDim.x + threadIdx.x;
  if (i < 10368) wsf[i] = 0.f;                       // sums, sumsq, cnt, csum
  if (i < 256)   wsf[WS_ZBUF + i] = 0.f;             // f16 zero pad (512 halves)
  if (i < NUMLOC * CC) {
    wsu[WS_LOCKEY + i] = NEGKEY;
    wsu[WS_SHPKEY + i] = NEGKEY;
  }
}

// ---------------- kernel 0a: NCHW f32 -> HWC f16 transpose (LDS-tiled) ----------------
__global__ __launch_bounds__(256) void pcc_xh(const float* __restrict__ xin, _Float16* __restrict__ xh) {
  __shared__ _Float16 tile[256 * 32];
  int blk  = blockIdx.x;                 // NB*HH*XTPR
  int bi   = blk / (HH * XTPR);
  int rem  = blk % (HH * XTPR);
  int row  = rem / XTPR;
  int c0   = (rem % XTPR) * 30;
  int t    = threadIdx.x;
  int colr = t & 31;                     // 0..31 (30 valid)
  int chg  = t >> 5;                     // 0..7
  const size_t HW = (size_t)HH * WW;
  const float* src = xin + (size_t)bi * CBEV * HW + (size_t)row * WW + c0;
#pragma unroll 4
  for (int it = 0; it < 32; ++it) {
    int ch = it * 8 + chg;
    float v = (colr < 30) ? src[(size_t)ch * HW + colr] : 0.f;
    tile[ch * 32 + colr] = (_Float16)v;
  }
  __syncthreads();
  _Float16* dst = xh + (((size_t)bi * HH + row) * WW + c0) * 256;
#pragma unroll 5
  for (int col = 0; col < 30; ++col) {
    dst[(size_t)col * 256 + t] = tile[t * 32 + col];
  }
}

// ---------------- kernel 0b: swizzle conv weights to f16 WMMA-B layout ----------------
// element index t = (((tap*8+q)*4+nt)*32+lane)*16+e
// B 32x16 f16 layout: lane -> N = nt*16 + (lane&15); k-in-chunk = (e>>3)*16 + (lane>>4)*8 + (e&7)
__global__ void pcc_bswz(const float* __restrict__ convw, _Float16* __restrict__ bswz) {
  int t = blockIdx.x * blockDim.x + threadIdx.x;
  if (t >= 9 * 8 * 4 * 32 * 16) return;
  int e    = t & 15;
  int lane = (t >> 4) & 31;
  int nt   = (t >> 9) & 3;
  int q    = (t >> 11) & 7;
  int tap  = t >> 14;
  int n  = nt * 16 + (lane & 15);
  int k  = q * 32 + (e >> 3) * 16 + (lane >> 4) * 8 + (e & 7);
  int ky = tap / 3, kx = tap % 3;
  float w = convw[(((size_t)n * CBEV + k) * 3 + ky) * 3 + kx];
  bswz[t] = (_Float16)w;
}

// ---------------- kernel 1: conv 3x3 via WMMA implicit GEMM + BN stats ----------------
__global__ __launch_bounds__(256) void pcc_conv(const _Float16* __restrict__ xh,
                                                const _Float16* __restrict__ zb,
                                                const _Float16* __restrict__ bswz,
                                                float* __restrict__ yh,
                                                float* __restrict__ stats) {
  __shared__ float sred[128];
  if (threadIdx.x < 128) sred[threadIdx.x] = 0.f;
  __syncthreads();

  const int lane = threadIdx.x & 31;
  const int tile = blockIdx.x * 8 + (threadIdx.x >> 5);
  const int bi   = tile / (HH * TPR);
  const int rem  = tile % (HH * TPR);
  const int h    = rem / TPR;
  const int x0t  = (rem % TPR) * 16;
  const int m    = lane & 15;
  const int hi   = lane >> 4;

  v8f acc0 = {}, acc1 = {}, acc2 = {}, acc3 = {};

  for (int tap = 0; tap < 9; ++tap) {
    int row = h + tap / 3 - 1;
    if (row < 0 || row >= HH) continue;            // wave-uniform
    int col  = x0t + m + (tap % 3) - 1;
    bool cv  = (col >= 0) && (col < WW);
    // branchless: OOB lanes read from the zero buffer
    const _Float16* xa = cv ? (xh + (((size_t)bi * HH + row) * WW + col) * 256) : zb;
    const _Float16* bt = bswz + ((size_t)tap * 1024 + lane) * 16;
    for (int q = 0; q < 8; ++q) {
      // A 16x32 f16 per-lane layout: two contiguous 16B chunks of channels
      h8 alo = *(const h8*)(xa + q * 32 + hi * 8);
      h8 ahi = *(const h8*)(xa + q * 32 + 16 + hi * 8);
      v16h a = __builtin_shufflevector(alo, ahi, 0, 1, 2, 3, 4, 5, 6, 7,
                                       8, 9, 10, 11, 12, 13, 14, 15);
      const _Float16* bp = bt + (size_t)q * 2048;
      v16h b0 = *(const v16h*)(bp);
      v16h b1 = *(const v16h*)(bp + 512);
      v16h b2 = *(const v16h*)(bp + 1024);
      v16h b3 = *(const v16h*)(bp + 1536);
      acc0 = __builtin_amdgcn_wmma_f32_16x16x32_f16(false, a, false, b0, (short)0, acc0, false, false);
      acc1 = __builtin_amdgcn_wmma_f32_16x16x32_f16(false, a, false, b1, (short)0, acc1, false, false);
      acc2 = __builtin_amdgcn_wmma_f32_16x16x32_f16(false, a, false, b2, (short)0, acc2, false, false);
      acc3 = __builtin_amdgcn_wmma_f32_16x16x32_f16(false, a, false, b3, (short)0, acc3, false, false);
    }
  }

  // C layout: VGPR r -> M = r + 8*hi, N = lane&15. Store HWC + BN partial sums.
  float s1a = 0, s1b = 0, s1c = 0, s1d = 0, s2a = 0, s2b = 0, s2c = 0, s2d = 0;
  size_t ob = (((size_t)bi * HH + h) * WW) * CC;
#pragma unroll
  for (int r = 0; r < 8; ++r) {
    int xo = x0t + r + hi * 8;
    if (xo < WW) {
      size_t o = ob + (size_t)xo * CC + (lane & 15);
      float v0 = acc0[r], v1 = acc1[r], v2 = acc2[r], v3 = acc3[r];
      yh[o] = v0; yh[o + 16] = v1; yh[o + 32] = v2; yh[o + 48] = v3;
      s1a += v0; s2a += v0 * v0;
      s1b += v1; s2b += v1 * v1;
      s1c += v2; s2c += v2 * v2;
      s1d += v3; s2d += v3 * v3;
    }
  }
  int ch = lane & 15;
  atomicAdd(&sred[ch], s1a);       atomicAdd(&sred[16 + ch], s1b);
  atomicAdd(&sred[32 + ch], s1c);  atomicAdd(&sred[48 + ch], s1d);
  atomicAdd(&sred[64 + ch], s2a);  atomicAdd(&sred[80 + ch], s2b);
  atomicAdd(&sred[96 + ch], s2c);  atomicAdd(&sred[112 + ch], s2d);
  __syncthreads();
  if (threadIdx.x < 128) atomicAdd(&stats[threadIdx.x], sred[threadIdx.x]);
}

// ---------------- kernel 2: BN affine fold ----------------
__global__ void pcc_bn(const float* __restrict__ g, const float* __restrict__ b, float* wsf) {
  int c = threadIdx.x;
  if (c >= CC) return;
  float cnt = (float)(NB * HH * WW);
  float mu  = wsf[WS_SUM + c] / cnt;
  float var = wsf[WS_SQ + c] / cnt - mu * mu;
  float sc  = g[c] * rsqrtf(var + 1e-5f);
  wsf[WS_SCALE + c] = sc;
  wsf[WS_SHIFT + c] = b[c] - mu * sc;
}

// ---------------- kernel 3: per-point gather + heads + segment atomics ----------------
__global__ __launch_bounds__(256) void pcc_points(const float* __restrict__ P,
                                                  const float* __restrict__ Y,
                                                  float* __restrict__ wsf, unsigned* __restrict__ wsu,
                                                  const float* __restrict__ wseg, const float* __restrict__ bseg,
                                                  const float* __restrict__ wreg, const float* __restrict__ breg,
                                                  const float* __restrict__ wemb, const float* __restrict__ bemb,
                                                  float* __restrict__ out) {
  __shared__ float sS[64], sH[64], sWs[192], sWr[192], sWe[128], sBs[3], sBr[3], sBe[2];
  int tid = threadIdx.x;
  for (int i = tid; i < 192; i += 256) { sWs[i] = wseg[i]; sWr[i] = wreg[i]; }
  for (int i = tid; i < 128; i += 256) { sWe[i] = wemb[i]; }
  if (tid < 64) { sS[tid] = wsf[WS_SCALE + tid]; sH[tid] = wsf[WS_SHIFT + tid]; }
  if (tid < 3)  { sBs[tid] = bseg[tid]; sBr[tid] = breg[tid]; }
  if (tid < 2)  { sBe[tid] = bemb[tid]; }
  __syncthreads();

  int p = blockIdx.x * 256 + tid;
  if (p >= NPTS) return;
  const float* pr = P + (size_t)p * 9;
  int pb = (int)pr[0];
  float x = pr[1], yy = pr[2], z = pr[3];
  int sw = (int)pr[6], in_ = (int)pr[7];

  const float CELL = (float)(0.075 * 8.0);
  float px = (x + 54.0f) / CELL;
  float py = (yy + 54.0f) / CELL;
  int ix0 = min(max((int)floorf(px), 0), WW - 1), ix1 = min(ix0 + 1, WW - 1);
  int iy0 = min(max((int)floorf(py), 0), HH - 1), iy1 = min(iy0 + 1, HH - 1);
  float wa = ((float)ix1 - px) * ((float)iy1 - py);
  float wb = ((float)ix1 - px) * (py - (float)iy0);
  float wc = (px - (float)ix0) * ((float)iy1 - py);
  float wd = (px - (float)ix0) * (py - (float)iy0);

  const float4* A4 = (const float4*)(Y + ((((size_t)pb * HH + iy0) * WW + ix0) << 6));
  const float4* B4 = (const float4*)(Y + ((((size_t)pb * HH + iy1) * WW + ix0) << 6));
  const float4* C4 = (const float4*)(Y + ((((size_t)pb * HH + iy0) * WW + ix1) << 6));
  const float4* D4 = (const float4*)(Y + ((((size_t)pb * HH + iy1) * WW + ix1) << 6));

  int seg = (pb * MAXI + in_) * SS + sw;
  unsigned* lk = wsu + WS_LOCKEY + (size_t)seg * 64;

  float a0 = sBs[0], a1 = sBs[1], a2 = sBs[2];
  float r0 = sBr[0], r1 = sBr[1], r2 = sBr[2];
  float e0 = sBe[0], e1 = sBe[1];

  for (int c4 = 0; c4 < 16; ++c4) {
    float4 fa = A4[c4], fb = B4[c4], fc = C4[c4], fd = D4[c4];
    const float* qa = (const float*)&fa;
    const float* qb = (const float*)&fb;
    const float* qc = (const float*)&fc;
    const float* qd = (const float*)&fd;
#pragma unroll
    for (int j = 0; j < 4; ++j) {
      int c = c4 * 4 + j;
      float sc = sS[c], sh = sH[c];
      float va = fmaxf(qa[j] * sc + sh, 0.f);
      float vb = fmaxf(qb[j] * sc + sh, 0.f);
      float vc = fmaxf(qc[j] * sc + sh, 0.f);
      float vd = fmaxf(qd[j] * sc + sh, 0.f);
      float v  = va * wa + vb * wb + vc * wc + vd * wd;
      a0 += v * sWs[c * 3 + 0]; a1 += v * sWs[c * 3 + 1]; a2 += v * sWs[c * 3 + 2];
      r0 += v * sWr[c * 3 + 0]; r1 += v * sWr[c * 3 + 1]; r2 += v * sWr[c * 3 + 2];
      e0 += v * sWe[c * 2 + 0]; e1 += v * sWe[c * 2 + 1];
      atomicMax(&lk[c], fkey(v));
    }
  }
  atomicAdd(&wsf[WS_CNT + seg], 1.f);
  atomicAdd(&wsf[WS_CSUM + seg * 3 + 0], x);
  atomicAdd(&wsf[WS_CSUM + seg * 3 + 1], yy);
  atomicAdd(&wsf[WS_CSUM + seg * 3 + 2], z);
  out[(size_t)p * 3 + 0] = a0; out[(size_t)p * 3 + 1] = a1; out[(size_t)p * 3 + 2] = a2;
  out[R_REG + (size_t)p * 3 + 0] = r0; out[R_REG + (size_t)p * 3 + 1] = r1; out[R_REG + (size_t)p * 3 + 2] = r2;
  out[R_EMB + (size_t)p * 2 + 0] = e0; out[R_EMB + (size_t)p * 2 + 1] = e1;
}

// ---------------- kernel 4: centroids ----------------
__global__ void pcc_centroid(float* wsf) {
  int i = blockIdx.x * blockDim.x + threadIdx.x;
  if (i < NUMLOC * 3) {
    int l = i / 3;
    wsf[WS_CENTROID + i] = wsf[WS_CSUM + i] / fmaxf(wsf[WS_CNT + l], 1.f);
  }
}

// ---------------- kernel 5: shape encoding (second point pass) ----------------
__global__ __launch_bounds__(256) void pcc_shape(const float* __restrict__ P,
                                                 const float* __restrict__ wsf, unsigned* __restrict__ wsu,
                                                 const float* __restrict__ wshape, const float* __restrict__ bshape) {
  __shared__ float sW[192], sB[64];
  int tid = threadIdx.x;
  for (int i = tid; i < 192; i += 256) sW[i] = wshape[i];
  if (tid < 64) sB[tid] = bshape[tid];
  __syncthreads();
  int p = blockIdx.x * 256 + tid;
  if (p >= NPTS) return;
  const float* pr = P + (size_t)p * 9;
  int pb = (int)pr[0];
  int sw = (int)pr[6], in_ = (int)pr[7];
  int seg = (pb * MAXI + in_) * SS + sw;
  float cx = pr[1] - wsf[WS_CENTROID + seg * 3 + 0];
  float cy = pr[2] - wsf[WS_CENTROID + seg * 3 + 1];
  float cz = pr[3] - wsf[WS_CENTROID + seg * 3 + 2];
  unsigned* sk = wsu + WS_SHPKEY + (size_t)seg * 64;
  for (int c = 0; c < 64; ++c) {
    float s = cx * sW[c] + cy * sW[64 + c] + cz * sW[128 + c] + sB[c];
    atomicMax(&sk[c], fkey(s));
  }
}

// ---------------- kernel 6: finalize locals_feat ----------------
__global__ void pcc_localsfin(float* wsf, const unsigned* wsu) {
  int i = blockIdx.x * blockDim.x + threadIdx.x;
  if (i < NUMLOC * CC) {
    int l = i >> 6;
    bool ne = wsf[WS_CNT + l] > 0.f;
    wsf[WS_LOCALS + i] = ne ? (fdec(wsu[WS_LOCKEY + i]) + fdec(wsu[WS_SHPKEY + i])) : 0.f;
  }
}

// ---------------- kernel 7: per-instance globals + inst_mos ----------------
__global__ void pcc_inst(float* wsf, const float* __restrict__ wmos, const float* __restrict__ bmos,
                         float* __restrict__ out) {
  __shared__ float part[64];
  int i = blockIdx.x, c = threadIdx.x;
  bool any = false;
  float g = -3.4e38f;
  for (int s = 0; s < SS; ++s) {
    if (wsf[WS_CNT + i * SS + s] > 0.f) {
      any = true;
      g = fmaxf(g, wsf[WS_LOCALS + ((size_t)(i * SS + s)) * 64 + c]);
    }
  }
  float gv = any ? g : 0.f;
  wsf[WS_GLOBALS + i * 64 + c] = gv;
  part[c] = gv * wmos[c];
  __syncthreads();
  if (c == 0) {
    int mx = -1, mn = SS;
    for (int s = 0; s < SS; ++s)
      if (wsf[WS_CNT + i * SS + s] > 0.f) { mx = s > mx ? s : mx; mn = s < mn ? s : mn; }
    int idmax = min(max(i * SS + mx, 0), NUMLOC - 1);
    int idmin = min(max(i * SS + mn, 0), NUMLOC - 1);
    float ic0 = wsf[WS_CENTROID + idmin * 3 + 0], ic1 = wsf[WS_CENTROID + idmin * 3 + 1], ic2 = wsf[WS_CENTROID + idmin * 3 + 2];
    float tc0 = wsf[WS_CENTROID + idmax * 3 + 0], tc1 = wsf[WS_CENTROID + idmax * 3 + 1], tc2 = wsf[WS_CENTROID + idmax * 3 + 2];
    wsf[WS_INITC + i * 3 + 0] = ic0; wsf[WS_INITC + i * 3 + 1] = ic1; wsf[WS_INITC + i * 3 + 2] = ic2;
    wsf[WS_TGTC + i * 3 + 0] = tc0; wsf[WS_TGTC + i * 3 + 1] = tc1; wsf[WS_TGTC + i * 3 + 2] = tc2;
    float acc = 0.f;
    for (int j = 0; j < 64; ++j) acc += part[j];
    acc += ic0 * wmos[64] + ic1 * wmos[65] + ic2 * wmos[66];
    acc += tc0 * wmos[67] + tc1 * wmos[68] + tc2 * wmos[69];
    out[R_MOS + i] = acc + bmos[0];
  }
}

// ---------------- kernel 8: locals_tf ----------------
__global__ void pcc_tf(const float* wsf, const float* __restrict__ wtf, const float* __restrict__ btf,
                       float* __restrict__ out) {
  int t = blockIdx.x * blockDim.x + threadIdx.x;
  if (t >= NUMLOC * 7) return;
  int l = t / 7, k = t % 7, inst = l / SS;
  float acc = btf[k];
  const float* lf = wsf + WS_LOCALS + (size_t)l * 64;
  const float* gf = wsf + WS_GLOBALS + (size_t)inst * 64;
  for (int r = 0; r < 64; ++r) acc += lf[r] * wtf[r * 7 + k];
  for (int r = 0; r < 64; ++r) acc += gf[r] * wtf[(64 + r) * 7 + k];
  acc += wsf[WS_CENTROID + l * 3 + 0] * wtf[128 * 7 + k];
  acc += wsf[WS_CENTROID + l * 3 + 1] * wtf[129 * 7 + k];
  acc += wsf[WS_CENTROID + l * 3 + 2] * wtf[130 * 7 + k];
  acc += wsf[WS_TGTC + inst * 3 + 0] * wtf[131 * 7 + k];
  acc += wsf[WS_TGTC + inst * 3 + 1] * wtf[132 * 7 + k];
  acc += wsf[WS_TGTC + inst * 3 + 2] * wtf[133 * 7 + k];
  out[R_TF + t] = acc;
}

extern "C" void kernel_launch(void* const* d_in, const int* in_sizes, int n_in,
                              void* d_out, int out_size, void* d_ws, size_t ws_size,
                              hipStream_t stream) {
  (void)in_sizes; (void)n_in; (void)out_size; (void)ws_size;
  const float* pts    = (const float*)d_in[0];
  const float* xin    = (const float*)d_in[1];
  const float* convw  = (const float*)d_in[2];
  const float* gamma  = (const float*)d_in[3];
  const float* beta   = (const float*)d_in[4];
  const float* wseg   = (const float*)d_in[5];
  const float* bseg   = (const float*)d_in[6];
  const float* wreg   = (const float*)d_in[7];
  const float* breg   = (const float*)d_in[8];
  const float* wemb   = (const float*)d_in[9];
  const float* bemb   = (const float*)d_in[10];
  const float* wshape = (const float*)d_in[11];
  const float* bshape = (const float*)d_in[12];
  const float* wtf    = (const float*)d_in[13];
  const float* btf    = (const float*)d_in[14];
  const float* wmos   = (const float*)d_in[15];
  const float* bmos   = (const float*)d_in[16];
  float*    out = (float*)d_out;
  float*    wsf = (float*)d_ws;
  unsigned* wsu = (unsigned*)d_ws;
  _Float16* bswz = (_Float16*)(wsf + WS_BSWZ);
  _Float16* xh   = (_Float16*)(wsf + WS_XH);
  const _Float16* zb = (const _Float16*)(wsf + WS_ZBUF);
  float*    yh   = wsf + WS_YHWC;

  pcc_init<<<(NUMLOC * CC + 255) / 256, 256, 0, stream>>>(wsf, wsu);
  pcc_xh<<<NB * HH * XTPR, 256, 0, stream>>>(xin, xh);
  pcc_bswz<<<(9 * 8 * 4 * 32 * 16 + 255) / 256, 256, 0, stream>>>(convw, bswz);
  pcc_conv<<<NTILES / 8, 256, 0, stream>>>(xh, zb, bswz, yh, wsf);
  pcc_bn<<<1, 64, 0, stream>>>(gamma, beta, wsf);
  pcc_points<<<(NPTS + 255) / 256, 256, 0, stream>>>(pts, yh, wsf, wsu,
                                                     wseg, bseg, wreg, breg, wemb, bemb, out);
  pcc_centroid<<<(NUMLOC * 3 + 255) / 256, 256, 0, stream>>>(wsf);
  pcc_shape<<<(NPTS + 255) / 256, 256, 0, stream>>>(pts, wsf, wsu, wshape, bshape);
  pcc_localsfin<<<(NUMLOC * CC + 255) / 256, 256, 0, stream>>>(wsf, wsu);
  pcc_inst<<<NUMINST, 64, 0, stream>>>(wsf, wmos, bmos, out);
  pcc_tf<<<(NUMLOC * 7 + 255) / 256, 256, 0, stream>>>(wsf, wtf, btf, out);
}
